// DKPF_Attention_7086696038468
// MI455X (gfx1250) — matrix operand order repaired and checked by reading, not compiled, not run on hardware
//
#include <hip/hip_runtime.h>
#include <hip/hip_bf16.h>

// ---------------------------------------------------------------------------
// Shapes: B=4, C=128, D=128, D4=32, H=W=64, HW=4096, N=B*HW=16384, K2=83
// ---------------------------------------------------------------------------

typedef __attribute__((ext_vector_type(16))) _Float16 v16h;
typedef __attribute__((ext_vector_type(8)))  float    v8f;

__device__ __forceinline__ v8f wmma_f16(v16h a, v16h b, v8f c) {
    // (neg_a, A, neg_b, B, c_mod, C, reuse_a, reuse_b)
    return __builtin_amdgcn_wmma_f32_16x16x32_f16(false, a, false, b, (short)0, c,
                                                  false, false);
}

// ---------------------------------------------------------------------------
// Kernel 0: re-layout de_w (32,128,3,3) OIHW -> f16 [o][tap][ci] so that the
// WMMA A-fragments of the tap-decomposed conv are contiguous 16B loads.
// ---------------------------------------------------------------------------
__global__ void k_wprep(const float* __restrict__ de_w, _Float16* __restrict__ wf) {
    int i   = blockIdx.x * 256 + threadIdx.x;   // 36864 = 32*9*128
    int ci  = i & 127;
    int ot  = i >> 7;                           // o*9 + tap
    int tap = ot % 9;
    int o   = ot / 9;
    wf[i] = (_Float16)de_w[(o * 128 + ci) * 9 + tap];
}

// ---------------------------------------------------------------------------
// Kernel 1: depth-encoder conv 3x3 (128 -> 32), tap-decomposed implicit GEMM.
//   Per tap (u,v): 1x1 GEMM with K=128 (4 WMMA steps) on shifted depth.
//   M = 32 (2 tiles), N = 16384 pixels (1024 tiles). 36 WMMAs per wave.
//   8 waves/block: wave gid -> mtile = gid&1, ntile = gid>>1. 256 blocks.
// ---------------------------------------------------------------------------
__global__ void k_conv_de(const float* __restrict__ depth,
                          const _Float16* __restrict__ wf,
                          const float* __restrict__ de_b,
                          float* __restrict__ outc) {
    const int wave = threadIdx.x >> 5, lane = threadIdx.x & 31;
    const int gid  = blockIdx.x * 8 + wave;
    const int m0   = (gid & 1) << 4;
    const int n0   = (gid >> 1) << 4;

    const int hi = lane >> 4;          // lane half
    const int nn = lane & 15;
    const int p  = n0 + nn;            // pixel for B/C columns
    const int pb = p >> 12;            // batch
    const int ph = (p >> 6) & 63;
    const int pw = p & 63;

    const int am  = m0 + nn;           // A row (out channel)
    const int akb = hi << 3;           // A K sub-base: 0 or 8
    const int bkb = hi << 4;           // B K base: 0 or 16
    const _Float16* __restrict__ arow = wf + am * 9 * 128;

    v8f acc = {};
    #pragma unroll
    for (int tap = 0; tap < 9; ++tap) {
        const int u = tap / 3, v = tap - (tap / 3) * 3;   // compile-time
        const int y = ph + u - 1, x = pw + v - 1;
        const bool valid = ((unsigned)y < 64u) && ((unsigned)x < 64u);
        #pragma unroll
        for (int kk = 0; kk < 128; kk += 32) {
            v16h afrag, bfrag;
            #pragma unroll
            for (int e = 0; e < 16; ++e) {
                int k = kk + ((e >> 3) << 4) + akb + (e & 7);
                afrag[e] = arow[tap * 128 + k];
            }
            if (valid) {
                const float* __restrict__ bsrc =
                    depth + ((pb * 128) << 12) + (y << 6) + x;
                #pragma unroll
                for (int e = 0; e < 16; ++e) {
                    int k = kk + ((e >> 3) << 4) + akb + (e & 7);
                    bfrag[e] = (_Float16)bsrc[k << 12];
                }
            } else {
                #pragma unroll
                for (int e = 0; e < 16; ++e) bfrag[e] = (_Float16)0.f;
            }
            acc = wmma_f16(afrag, bfrag, acc);
        }
    }
    const int mb = m0 + (hi << 3);
    #pragma unroll
    for (int r = 0; r < 8; ++r) {
        int o = mb + r;
        outc[((pb * 32 + o) << 12) + (ph << 6) + pw] = acc[r] + de_b[o];
    }
}

// ---------------------------------------------------------------------------
// Kernel 2: BN statistics per channel (mean, rsqrt(var+eps)).
// ---------------------------------------------------------------------------
__global__ void k_bnstats(const float* __restrict__ x, float* __restrict__ stats) {
    __shared__ float s1[256], s2[256];
    const int c = blockIdx.x;          // 0..31
    float sum = 0.f, sq = 0.f;
    for (int i = threadIdx.x; i < 4 * 4096; i += 256) {
        int b = i >> 12, hw = i & 4095;
        float v = x[((b * 32 + c) << 12) + hw];
        sum += v; sq += v * v;
    }
    s1[threadIdx.x] = sum; s2[threadIdx.x] = sq;
    __syncthreads();
    for (int s = 128; s > 0; s >>= 1) {
        if (threadIdx.x < s) {
            s1[threadIdx.x] += s1[threadIdx.x + s];
            s2[threadIdx.x] += s2[threadIdx.x + s];
        }
        __syncthreads();
    }
    if (threadIdx.x == 0) {
        float m   = s1[0] * (1.f / 16384.f);
        float var = s2[0] * (1.f / 16384.f) - m * m;
        stats[c * 2]     = m;
        stats[c * 2 + 1] = rsqrtf(var + 1e-5f);
    }
}

// ---------------------------------------------------------------------------
// Kernel 3: BN apply + ReLU  (524288 elements)
// ---------------------------------------------------------------------------
__global__ void k_bnapply(const float* __restrict__ x, const float* __restrict__ stats,
                          const float* __restrict__ g, const float* __restrict__ bb,
                          float* __restrict__ enc) {
    int idx = blockIdx.x * 256 + threadIdx.x;
    int c = (idx >> 12) & 31;
    float v = (x[idx] - stats[c * 2]) * stats[c * 2 + 1] * g[c] + bb[c];
    enc[idx] = v > 0.f ? v : 0.f;
}

// ---------------------------------------------------------------------------
// Kernel 4: kernel-generator 1x1 conv (32 -> 83) via single WMMA.
//   M = 83 (6 tiles padded to 96), K = 32 (one step), N = 16384 (1024 tiles).
//   6144 waves -> 768 blocks * 8 waves.
// ---------------------------------------------------------------------------
__global__ void k_kwgen(const float* __restrict__ enc, const float* __restrict__ kg_w,
                        const float* __restrict__ kg_b, float* __restrict__ kwbuf) {
    const int wave = threadIdx.x >> 5, lane = threadIdx.x & 31;
    const int gid  = blockIdx.x * 8 + wave;
    const int m0   = (gid % 6) << 4;
    const int n0   = (gid / 6) << 4;
    const int hi = lane >> 4, nn = lane & 15;
    const int p = n0 + nn, pb = p >> 12, hw = p & 4095;

    const int am = m0 + nn, akb = hi << 3, bkb = hi << 4;
    v16h afrag, bfrag;
    #pragma unroll
    for (int e = 0; e < 16; ++e) {
        int k = ((e >> 3) << 4) + akb + (e & 7);
        afrag[e] = (_Float16)((am < 83) ? kg_w[am * 32 + k] : 0.f);
    }
    #pragma unroll
    for (int e = 0; e < 16; ++e) {
        int k = ((e >> 3) << 4) + akb + (e & 7);
        bfrag[e] = (_Float16)enc[((pb * 32 + k) << 12) + hw];
    }
    v8f acc = {};
    acc = wmma_f16(afrag, bfrag, acc);
    const int mb = m0 + (hi << 3);
    #pragma unroll
    for (int r = 0; r < 8; ++r) {
        int o = mb + r;
        if (o < 83)
            kwbuf[((pb * 83 + o) << 12) + hw] = acc[r] + kg_b[o];
    }
}

// ---------------------------------------------------------------------------
// Kernel 5: gate conv 3x3 (256 -> 3) + channel softmax. Thread per pixel.
// ---------------------------------------------------------------------------
__global__ void k_gates(const float* __restrict__ rgb, const float* __restrict__ depth,
                        const float* __restrict__ gw, const float* __restrict__ gb,
                        float* __restrict__ gates) {
    int p = blockIdx.x * 256 + threadIdx.x;           // 16384
    int b = p >> 12, h = (p >> 6) & 63, w = p & 63;
    float z0 = gb[0], z1 = gb[1], z2 = gb[2];
    for (int ci = 0; ci < 256; ++ci) {
        const float* src = (ci < 128) ? (rgb   + ((b * 128 + ci) << 12))
                                      : (depth + ((b * 128 + (ci - 128)) << 12));
        float t[9];
        #pragma unroll
        for (int u = 0; u < 3; ++u)
            #pragma unroll
            for (int v = 0; v < 3; ++v) {
                int y = h + u - 1, x = w + v - 1;
                t[u * 3 + v] = ((unsigned)y < 64u && (unsigned)x < 64u)
                                   ? src[(y << 6) + x] : 0.f;
            }
        #pragma unroll
        for (int r = 0; r < 9; ++r) {
            z0 += t[r] * gw[(0 * 256 + ci) * 9 + r];
            z1 += t[r] * gw[(1 * 256 + ci) * 9 + r];
            z2 += t[r] * gw[(2 * 256 + ci) * 9 + r];
        }
    }
    float m  = fmaxf(z0, fmaxf(z1, z2));
    float e0 = __expf(z0 - m), e1 = __expf(z1 - m), e2 = __expf(z2 - m);
    float inv = 1.f / (e0 + e1 + e2);
    int hw = p & 4095;
    gates[((b * 3 + 0) << 12) + hw] = e0 * inv;
    gates[((b * 3 + 1) << 12) + hw] = e1 * inv;
    gates[((b * 3 + 2) << 12) + hw] = e2 * inv;
}

// ---------------------------------------------------------------------------
// Kernel 6: global mean over H,W of concat(rgb, depth) -> gm[b][256]
// ---------------------------------------------------------------------------
__global__ void k_gmean(const float* __restrict__ rgb, const float* __restrict__ depth,
                        float* __restrict__ gm) {
    __shared__ float s[256];
    int b  = blockIdx.x >> 8;
    int ci = blockIdx.x & 255;
    const float* src = (ci < 128) ? rgb + ((b * 128 + ci) << 12)
                                  : depth + ((b * 128 + (ci - 128)) << 12);
    float sum = 0.f;
    for (int i = threadIdx.x; i < 4096; i += 256) sum += src[i];
    s[threadIdx.x] = sum;
    __syncthreads();
    for (int st = 128; st > 0; st >>= 1) {
        if (threadIdx.x < st) s[threadIdx.x] += s[threadIdx.x + st];
        __syncthreads();
    }
    if (threadIdx.x == 0) gm[b * 256 + ci] = s[0] * (1.f / 4096.f);
}

// ---------------------------------------------------------------------------
// Kernel 7: attention head — a1 (3x256) + ReLU, a2 (3x3), softmax. 1 block.
// ---------------------------------------------------------------------------
__global__ void k_att(const float* __restrict__ gm,
                      const float* __restrict__ a1w, const float* __restrict__ a1b,
                      const float* __restrict__ a2w, const float* __restrict__ a2b,
                      float* __restrict__ att) {
    __shared__ float t[12];
    int tid = threadIdx.x;
    if (tid < 12) {
        int b = tid / 3, a = tid % 3;
        float s = a1b[a];
        for (int ci = 0; ci < 256; ++ci) s += gm[b * 256 + ci] * a1w[a * 256 + ci];
        t[tid] = s > 0.f ? s : 0.f;
    }
    __syncthreads();
    if (tid < 4) {
        int b = tid;
        float z[3];
        #pragma unroll
        for (int a = 0; a < 3; ++a) {
            z[a] = a2b[a];
            #pragma unroll
            for (int j = 0; j < 3; ++j) z[a] += t[b * 3 + j] * a2w[a * 3 + j];
        }
        float m  = fmaxf(z[0], fmaxf(z[1], z[2]));
        float e0 = __expf(z[0] - m), e1 = __expf(z[1] - m), e2 = __expf(z[2] - m);
        float inv = 1.f / (e0 + e1 + e2);
        att[b * 3 + 0] = e0 * inv;
        att[b * 3 + 1] = e1 * inv;
        att[b * 3 + 2] = e2 * inv;
    }
}

// ---------------------------------------------------------------------------
// Kernel 8: depthwise kxk conv on rgb -> f. Thread per output element.
// ---------------------------------------------------------------------------
__global__ void k_dw(const float* __restrict__ rgb, const float* __restrict__ wt,
                     const float* __restrict__ bias, float* __restrict__ f, int k) {
    int idx = blockIdx.x * 256 + threadIdx.x;
    int bc  = idx >> 12;
    int c   = bc & 127;
    int h   = (idx >> 6) & 63, w = idx & 63;
    int pd  = k >> 1;
    const float* src = rgb + (bc << 12);
    const float* wc  = wt + c * k * k;
    // Exercise the gfx1250 prefetch path for the next row of this plane.
    __builtin_prefetch(src + ((((h + 1) & 63)) << 6), 0, 0);
    float acc = bias[c];
    for (int u = 0; u < k; ++u) {
        int y = h + u - pd;
        if ((unsigned)y >= 64u) continue;
        const float* row = src + (y << 6);
        for (int v = 0; v < k; ++v) {
            int x = w + v - pd;
            if ((unsigned)x < 64u) acc += row[x] * wc[u * k + v];
        }
    }
    f[idx] = acc;
}

// ---------------------------------------------------------------------------
// Kernel 9: dynamic per-pixel conv of f with kwbuf slice, scaled by
// gates[:,ai]*att[:,ai], accumulated into out (first pass overwrites).
// ---------------------------------------------------------------------------
__global__ void k_dyn(const float* __restrict__ f, const float* __restrict__ kwbuf,
                      const float* __restrict__ gates, const float* __restrict__ att,
                      float* __restrict__ out, int k, int koff, int ai, int first) {
    int idx = blockIdx.x * 256 + threadIdx.x;
    int bc  = idx >> 12;
    int b   = bc >> 7;
    int hw  = idx & 4095;
    int h   = hw >> 6, w = hw & 63;
    int pd  = k >> 1;
    const float* src = f + (bc << 12);
    const float* kp  = kwbuf + ((b * 83 + koff) << 12) + hw;   // stride 4096 per tap
    float acc = 0.f;
    int t = 0;
    for (int u = 0; u < k; ++u) {
        int y = h + u - pd;
        for (int v = 0; v < k; ++v, ++t) {
            int x = w + v - pd;
            if ((unsigned)y < 64u && (unsigned)x < 64u)
                acc += src[(y << 6) + x] * kp[t << 12];
        }
    }
    float gsc = gates[((b * 3 + ai) << 12) + hw] * att[b * 3 + ai];
    float r = gsc * acc;
    if (first) out[idx] = r;
    else       out[idx] += r;
}

// ---------------------------------------------------------------------------
// Kernel 10: residual 1x1 conv (128 -> 128) via WMMA with LDS-staged B tile.
//   Block = 8 waves = all 8 M tiles of one 16-pixel N tile (ntile = blockIdx).
//   B tile (K=128 x N=16, f16) staged once in LDS, shared by all waves.
// ---------------------------------------------------------------------------
__global__ void k_res(const float* __restrict__ rgb, const float* __restrict__ rw,
                      const float* __restrict__ rb, float* __restrict__ out) {
    __shared__ _Float16 Bs[16][136];   // [n][k], padded row to de-phase banks
    const int tid = threadIdx.x;
    const int n0  = blockIdx.x << 4;

    // Cooperative stage: 2048 elements, 8 per thread, convert to f16.
    for (int i = tid; i < 16 * 128; i += 256) {
        int n = i >> 7, k = i & 127;
        int p = n0 + n, pb = p >> 12, hw = p & 4095;
        Bs[n][k] = (_Float16)rgb[((pb * 128 + k) << 12) + hw];
    }
    __syncthreads();

    const int wave = tid >> 5, lane = tid & 31;
    const int m0 = wave << 4;
    const int hi = lane >> 4, nn = lane & 15;
    const int p = n0 + nn, pb = p >> 12, hw = p & 4095;
    const int am = m0 + nn, akb = hi << 3, bkb = hi << 4;

    v8f acc = {};
    #pragma unroll
    for (int kk = 0; kk < 128; kk += 32) {
        v16h afrag, bfrag;
        #pragma unroll
        for (int e = 0; e < 16; ++e) {
            int k = kk + ((e >> 3) << 4) + akb + (e & 7);
            afrag[e] = (_Float16)rw[am * 128 + k];
        }
        #pragma unroll
        for (int e = 0; e < 16; ++e) {
            int k = kk + ((e >> 3) << 4) + akb + (e & 7);
            bfrag[e] = Bs[nn][k];
        }
        acc = wmma_f16(afrag, bfrag, acc);
    }
    const int mb = m0 + (hi << 3);
    #pragma unroll
    for (int r = 0; r < 8; ++r) {
        int o  = mb + r;
        int oi = ((pb * 128 + o) << 12) + hw;
        out[oi] = out[oi] + acc[r] + rb[o];
    }
}

// ---------------------------------------------------------------------------
// Host launcher
// ---------------------------------------------------------------------------
extern "C" void kernel_launch(void* const* d_in, const int* in_sizes, int n_in,
                              void* d_out, int out_size, void* d_ws, size_t ws_size,
                              hipStream_t stream) {
    (void)in_sizes; (void)n_in; (void)out_size; (void)ws_size;
    const float* rgb    = (const float*)d_in[0];
    const float* depth  = (const float*)d_in[1];
    const float* de_w   = (const float*)d_in[2];
    const float* de_b   = (const float*)d_in[3];
    const float* bn_g   = (const float*)d_in[4];
    const float* bn_b   = (const float*)d_in[5];
    const float* kg_w   = (const float*)d_in[6];
    const float* kg_b   = (const float*)d_in[7];
    const float* w3     = (const float*)d_in[8];
    const float* b3     = (const float*)d_in[9];
    const float* w5     = (const float*)d_in[10];
    const float* b5     = (const float*)d_in[11];
    const float* w7     = (const float*)d_in[12];
    const float* b7     = (const float*)d_in[13];
    const float* gate_w = (const float*)d_in[14];
    const float* gate_b = (const float*)d_in[15];
    const float* a1_w   = (const float*)d_in[16];
    const float* a1_b   = (const float*)d_in[17];
    const float* a2_w   = (const float*)d_in[18];
    const float* a2_b   = (const float*)d_in[19];
    const float* res_w  = (const float*)d_in[20];
    const float* res_b  = (const float*)d_in[21];
    float* out = (float*)d_out;

    float* ws    = (float*)d_ws;
    float* conv1 = ws;                 // 524288  : raw de-conv output
    float* stats = ws + 524288;        // 64      : per-ch mean / inv-std
    float* enc   = ws + 524352;        // 524288  : BN+ReLU depth encoding
    float* kwbuf = ws + 1048640;       // 1359872 : dynamic kernels (B,83,H,W)
    float* gates = ws + 2408512;       // 49152   : softmax gates (B,3,H,W)
    float* gm    = ws + 2457664;       // 1024    : global means (B,256)
    float* att   = ws + 2458688;       // 12 (padded to 64)
    float* fbuf  = ws + 2458752;       // 2097152 : depthwise output (reused per k)
    _Float16* wf = (_Float16*)(ws + 4555904); // 36864 f16 : transposed de_w
                                       // total  = 4574336 floats (~18.3 MB)

    k_wprep  <<<144,  256, 0, stream>>>(de_w, wf);
    k_conv_de<<<256,  256, 0, stream>>>(depth, wf, de_b, conv1);
    k_bnstats<<<32,   256, 0, stream>>>(conv1, stats);
    k_bnapply<<<2048, 256, 0, stream>>>(conv1, stats, bn_g, bn_b, enc);
    k_kwgen  <<<768,  256, 0, stream>>>(enc, kg_w, kg_b, kwbuf);
    k_gates  <<<64,   256, 0, stream>>>(rgb, depth, gate_w, gate_b, gates);
    k_gmean  <<<1024, 256, 0, stream>>>(rgb, depth, gm);
    k_att    <<<1,     32, 0, stream>>>(gm, a1_w, a1_b, a2_w, a2_b, att);

    // k = 3 (kernel slots 0..8)
    k_dw <<<8192, 256, 0, stream>>>(rgb, w3, b3, fbuf, 3);
    k_dyn<<<8192, 256, 0, stream>>>(fbuf, kwbuf, gates, att, out, 3, 0, 0, 1);
    // k = 5 (slots 9..33)
    k_dw <<<8192, 256, 0, stream>>>(rgb, w5, b5, fbuf, 5);
    k_dyn<<<8192, 256, 0, stream>>>(fbuf, kwbuf, gates, att, out, 5, 9, 1, 0);
    // k = 7 (slots 34..82)
    k_dw <<<8192, 256, 0, stream>>>(rgb, w7, b7, fbuf, 7);
    k_dyn<<<8192, 256, 0, stream>>>(fbuf, kwbuf, gates, att, out, 7, 34, 2, 0);

    // residual 1x1 conv added on top of fused result
    k_res<<<1024, 256, 0, stream>>>(rgb, res_w, res_b, out);
}
